// MotionGPT_8796093022601
// MI455X (gfx1250) — matrix operand order repaired and checked by reading, not compiled, not run on hardware
//
#include <hip/hip_runtime.h>
#include <math.h>
#include <stdint.h>

// ---------------- constants ----------------
#define BB   64
#define TT   255
#define DD   512
#define HH   8
#define LL   8
#define VV   515
#define SS   256          // T + 1 (cond token prepended)
#define MROWS (BB * SS)   // 16384 token rows
#define HD   64

typedef __attribute__((ext_vector_type(16))) __bf16 v16bf;
typedef __attribute__((ext_vector_type(8)))  float  v8f;

static __device__ __forceinline__ unsigned short f2bf(float f) {
    return __builtin_bit_cast(unsigned short, (__bf16)f);
}
static __device__ __forceinline__ unsigned int pack2bf(float lo, float hi) {
    return (unsigned int)f2bf(lo) | ((unsigned int)f2bf(hi) << 16);
}

// ---------------- weight convert + transpose: [L][K][N] f32 -> [L][N][K] bf16 ----------------
__global__ void k_convT(const float* __restrict__ in, unsigned short* __restrict__ out,
                        int K, int N, int total) {
    int i = blockIdx.x * blockDim.x + threadIdx.x;
    if (i >= total) return;
    int nk = N * K;
    int l = i / nk;
    int rem = i - l * nk;
    int n = rem / K;
    int k = rem - n * K;
    out[i] = f2bf(in[(size_t)l * nk + (size_t)k * N + n]);
}

// emb [V,D] is already [N][K]: straight convert with zero-padded rows to Npad
__global__ void k_conv_pad(const float* __restrict__ in, unsigned short* __restrict__ out,
                           int K, int Nreal, int total) {
    int i = blockIdx.x * blockDim.x + threadIdx.x;
    if (i >= total) return;
    int n = i / K;
    int k = i - n * K;
    out[i] = (n < Nreal) ? f2bf(in[(size_t)n * K + k]) : (unsigned short)0;
}

// cond [64,512] -> padded [256,512] (zero rows 64..255)
__global__ void k_pad_cond(const float* __restrict__ in, float* __restrict__ out) {
    int i = blockIdx.x * blockDim.x + threadIdx.x;
    if (i >= 256 * DD) return;
    int row = i >> 9;
    out[i] = (row < BB) ? in[i] : 0.0f;
}

// ---------------- embedding + cond concat ----------------
__global__ void k_embed(const int* __restrict__ tokens,
                        const float* __restrict__ emb,
                        const float* __restrict__ pos,
                        const float* __restrict__ cond_out,
                        float* __restrict__ h) {
    int i = blockIdx.x * blockDim.x + threadIdx.x;
    if (i >= MROWS * DD) return;
    int d   = i & (DD - 1);
    int row = i >> 9;
    int s   = row & (SS - 1);
    int b   = row >> 8;
    if (s == 0) {
        h[i] = cond_out[b * DD + d];
    } else {
        int t   = s - 1;
        int tok = tokens[b * TT + t];
        h[i] = emb[(size_t)tok * DD + d] + pos[(size_t)t * DD + d];
    }
}

// ---------------- layernorm (one row per block, D=512) ----------------
__global__ void k_layernorm(const float* __restrict__ in,
                            const float* __restrict__ g,
                            const float* __restrict__ bta,
                            float* __restrict__ out) {
    __shared__ float red[256];
    const int row = blockIdx.x;
    const int tid = threadIdx.x;
    const float* p = in + (size_t)row * DD;
    float a0 = p[tid], a1 = p[tid + 256];
    red[tid] = a0 + a1;
    __syncthreads();
    for (int s = 128; s > 0; s >>= 1) {
        if (tid < s) red[tid] += red[tid + s];
        __syncthreads();
    }
    float mean = red[0] * (1.0f / DD);
    __syncthreads();
    float d0 = a0 - mean, d1 = a1 - mean;
    red[tid] = d0 * d0 + d1 * d1;
    __syncthreads();
    for (int s = 128; s > 0; s >>= 1) {
        if (tid < s) red[tid] += red[tid + s];
        __syncthreads();
    }
    float r = rsqrtf(red[0] * (1.0f / DD) + 1e-5f);
    float* o = out + (size_t)row * DD;
    o[tid]       = d0 * r * g[tid]       + bta[tid];
    o[tid + 256] = d1 * r * g[tid + 256] + bta[tid + 256];
}

// ---------------- WMMA GEMM ----------------
// C[M,N] = act(A[M,K] @ W^T + bias) (+resid).  W passed TRANSPOSED bf16 [Npad][K].
// Block = 256 threads (8 waves). Block tile 256(M) x 64(N), K step 32.
// Each wave computes 32x64: 8 x v_wmma_f32_16x16x32_bf16 per K step (B frags reused).
// A rows [m0, m0+255] must be readable (buffers padded); W rows [n0, n0+63] readable.
#define ACT_NONE 0
#define ACT_SILU 1
#define ACT_GELU 2

__global__ __launch_bounds__(256)
void k_gemm_bf16(const float* __restrict__ A,
                 const unsigned short* __restrict__ WT,   // [Npad][K] bf16
                 const float* __restrict__ bias,
                 const float* __restrict__ resid,
                 float* __restrict__ C,
                 int M, int N, int K, int act, int logits_mode) {
    __shared__ unsigned short As[256 * 32];   // [m][k]
    __shared__ unsigned short Bs[64 * 32];    // [n][k]

    const int tid  = threadIdx.x;
    const int lane = tid & 31;
    const int wid  = tid >> 5;
    const int m0   = blockIdx.y * 256;
    const int n0   = blockIdx.x * 64;

    union Frag { unsigned int u[8]; uint4 q[2]; v16bf v; };

    v8f acc[2][4];
#pragma unroll
    for (int s = 0; s < 2; ++s)
#pragma unroll
        for (int j = 0; j < 4; ++j)
#pragma unroll
            for (int r = 0; r < 8; ++r) acc[s][j][r] = 0.0f;

    const int rA0 = wid * 32 + (lane & 15);          // first A-frag row (local)
    const int kgA = (lane & 16) ? 8 : 0;
    const int kgB = (lane & 16) ? 8 : 0;

    // B staging: each thread owns one contiguous 16B chunk (8 bf16) per K step
    const int bRow = tid >> 2;                       // 0..63
    const int bC8  = (tid & 3) * 8;                  // 0/8/16/24
    const unsigned short* bSrc = WT + (size_t)(n0 + bRow) * K + bC8;
    const unsigned bDstLds = (unsigned)(uintptr_t)(&Bs[bRow * 32 + bC8]);

    for (int kb = 0; kb < K; kb += 32) {
        // --- B tile: async global -> LDS (16B per thread, ASYNCcnt-tracked) ---
        {
            unsigned long long src64 = (unsigned long long)(uintptr_t)(bSrc + kb);
            asm volatile("global_load_async_to_lds_b128 %0, %1, off"
                         :: "v"(bDstLds), "v"(src64) : "memory");
        }
        // --- A tile: 256x32 fp32 -> bf16 (32 elems / thread, 16B vector ops) ---
#pragma unroll
        for (int t = 0; t < 4; ++t) {
            int idx8 = t * 256 + tid;                // 0..1023 chunks of 8
            int r  = idx8 >> 2;
            int c8 = (idx8 & 3) * 8;
            const float* src = A + (size_t)(m0 + r) * K + kb + c8;
            float4 v0 = *(const float4*)(src);
            float4 v1 = *(const float4*)(src + 4);
            uint4 pk;
            pk.x = pack2bf(v0.x, v0.y);
            pk.y = pack2bf(v0.z, v0.w);
            pk.z = pack2bf(v1.x, v1.y);
            pk.w = pack2bf(v1.z, v1.w);
            *(uint4*)&As[r * 32 + c8] = pk;
        }
        asm volatile("s_wait_asynccnt 0" ::: "memory");
        __syncthreads();

        // --- fragments + WMMA ---
        Frag a0, a1;
        a0.q[0] = *(const uint4*)&As[rA0 * 32 + kgA];
        a0.q[1] = *(const uint4*)&As[rA0 * 32 + kgA + 16];
        a1.q[0] = *(const uint4*)&As[(rA0 + 16) * 32 + kgA];
        a1.q[1] = *(const uint4*)&As[(rA0 + 16) * 32 + kgA + 16];
#pragma unroll
        for (int j = 0; j < 4; ++j) {
            Frag bf;
            int nl = j * 16 + (lane & 15);
            bf.q[0] = *(const uint4*)&Bs[nl * 32 + kgB];
            bf.q[1] = *(const uint4*)&Bs[nl * 32 + kgB + 16];
            acc[0][j] = __builtin_amdgcn_wmma_f32_16x16x32_bf16(
                false, a0.v, false, bf.v, (short)0, acc[0][j], false, false);
            acc[1][j] = __builtin_amdgcn_wmma_f32_16x16x32_bf16(
                false, a1.v, false, bf.v, (short)0, acc[1][j], false, false);
        }
        __syncthreads();
    }

    // --- epilogue: bias + activation + residual + store ---
#pragma unroll
    for (int s = 0; s < 2; ++s) {
        const int mb = m0 + wid * 32 + s * 16 + ((lane & 16) ? 8 : 0);
#pragma unroll
        for (int j = 0; j < 4; ++j) {
            int nn = n0 + j * 16 + (lane & 15);
            if (nn >= N) continue;
#pragma unroll
            for (int r = 0; r < 8; ++r) {
                int mm = mb + r;
                if (mm >= M) continue;
                float v = acc[s][j][r];
                if (bias) v += bias[nn];
                if (act == ACT_SILU) {
                    v = v / (1.0f + __expf(-v));
                } else if (act == ACT_GELU) {
                    v = 0.5f * v * (1.0f + erff(v * 0.70710678118654752f));
                }
                if (resid) v += resid[(size_t)mm * N + nn];
                if (logits_mode) {
                    int b = mm >> 8, sidx = mm & (SS - 1);
                    if (sidx == 0) continue;          // drop cond row
                    C[((size_t)b * TT + (sidx - 1)) * N + nn] = v;
                } else {
                    C[(size_t)mm * N + nn] = v;
                }
            }
        }
    }
}

// ---------------- causal attention, one (batch, head) per block ----------------
__global__ __launch_bounds__(256)
void k_attention(const float* __restrict__ qkv,  // [M, 3*D]
                 float* __restrict__ out) {      // [M, D]
    __shared__ float Ks[64][64];
    __shared__ float Vs[64][64];

    const int bh   = blockIdx.x;
    const int b    = bh >> 3;
    const int head = bh & 7;
    const int s    = threadIdx.x;   // query row
    const float scale = 0.125f;

    const size_t rowbase = (size_t)(b * SS) * (3 * DD);

    float q[HD];
    {
        const float* qp = qkv + rowbase + (size_t)s * (3 * DD) + head * HD;
#pragma unroll
        for (int d = 0; d < HD; ++d) q[d] = qp[d];
    }

    float o[HD];
#pragma unroll
    for (int d = 0; d < HD; ++d) o[d] = 0.0f;
    float mrun = -3.0e38f;
    float lrun = 0.0f;

    for (int c0 = 0; c0 < SS; c0 += 64) {
#pragma unroll
        for (int t = 0; t < 16; ++t) {
            int idx = t * 256 + threadIdx.x;
            int r = idx >> 6, d = idx & 63;
            size_t src = rowbase + (size_t)(c0 + r) * (3 * DD) + head * HD + d;
            Ks[r][d] = qkv[src + DD];
            Vs[r][d] = qkv[src + 2 * DD];
        }
        __syncthreads();

        int kmax = (s - c0 < 63) ? (s - c0) : 63;
        for (int kk = 0; kk <= kmax; ++kk) {
            float dot = 0.0f;
#pragma unroll
            for (int d = 0; d < HD; ++d) dot += q[d] * Ks[kk][d];
            float xsc = dot * scale;
            float mnew = fmaxf(mrun, xsc);
            float corr = __expf(mrun - mnew);
            float p    = __expf(xsc - mnew);
            lrun = lrun * corr + p;
#pragma unroll
            for (int d = 0; d < HD; ++d) o[d] = o[d] * corr + p * Vs[kk][d];
            mrun = mnew;
        }
        __syncthreads();
    }

    float inv = 1.0f / lrun;
    float* op = out + (size_t)(b * SS + s) * DD + head * HD;
#pragma unroll
    for (int d = 0; d < HD; ++d) op[d] = o[d] * inv;
}

// ---------------- host launch ----------------
extern "C" void kernel_launch(void* const* d_in, const int* in_sizes, int n_in,
                              void* d_out, int out_size, void* d_ws, size_t ws_size,
                              hipStream_t stream) {
    (void)in_sizes; (void)n_in; (void)out_size; (void)ws_size;

    const int*   tokens  = (const int*)  d_in[0];
    const float* cond    = (const float*)d_in[1];
    const float* emb     = (const float*)d_in[2];
    const float* pos_emb = (const float*)d_in[3];
    const float* cond_w1 = (const float*)d_in[4];
    const float* cond_b1 = (const float*)d_in[5];
    const float* cond_w2 = (const float*)d_in[6];
    const float* cond_b2 = (const float*)d_in[7];
    const float* ln1_g   = (const float*)d_in[8];
    const float* ln1_b   = (const float*)d_in[9];
    const float* qkv_w   = (const float*)d_in[10];
    const float* qkv_b   = (const float*)d_in[11];
    const float* proj_w  = (const float*)d_in[12];
    const float* proj_b  = (const float*)d_in[13];
    const float* ln2_g   = (const float*)d_in[14];
    const float* ln2_b   = (const float*)d_in[15];
    const float* mlp_w1  = (const float*)d_in[16];
    const float* mlp_b1  = (const float*)d_in[17];
    const float* mlp_w2  = (const float*)d_in[18];
    const float* mlp_b2  = (const float*)d_in[19];
    const float* lnf_g   = (const float*)d_in[20];
    const float* lnf_b   = (const float*)d_in[21];
    float* outp = (float*)d_out;

    char* ws = (char*)d_ws;
    size_t off = 0;
    auto alloc = [&](size_t bytes) -> void* {
        void* p = ws + off;
        off = (off + bytes + 255) & ~(size_t)255;
        return p;
    };

    const int VPAD = 576;  // ceil(515/64)*64
    unsigned short* qkvT  = (unsigned short*)alloc((size_t)LL * 3 * DD * DD * 2);
    unsigned short* projT = (unsigned short*)alloc((size_t)LL * DD * DD * 2);
    unsigned short* m1T   = (unsigned short*)alloc((size_t)LL * 4 * DD * DD * 2);
    unsigned short* m2T   = (unsigned short*)alloc((size_t)LL * DD * 4 * DD * 2);
    unsigned short* cw1T  = (unsigned short*)alloc((size_t)DD * DD * 2);
    unsigned short* cw2T  = (unsigned short*)alloc((size_t)DD * DD * 2);
    unsigned short* lgT   = (unsigned short*)alloc((size_t)VPAD * DD * 2);
    float* h     = (float*)alloc((size_t)MROWS * DD * 4);
    float* x     = (float*)alloc((size_t)MROWS * DD * 4);
    float* big   = (float*)alloc((size_t)MROWS * 4 * DD * 4);  // qkv / mlp hidden
    float* attn  = (float*)alloc((size_t)MROWS * DD * 4);
    float* condp = (float*)alloc((size_t)256 * DD * 4);        // padded cond
    float* chid  = (float*)alloc((size_t)256 * DD * 4);
    float* cout  = (float*)alloc((size_t)256 * DD * 4);

    auto convT = [&](const float* src, unsigned short* dst, int Lc, int K, int N) {
        int total = Lc * N * K;
        k_convT<<<(total + 255) / 256, 256, 0, stream>>>(src, dst, K, N, total);
    };
    convT(qkv_w,   qkvT,  LL, DD,     3 * DD);
    convT(proj_w,  projT, LL, DD,     DD);
    convT(mlp_w1,  m1T,   LL, DD,     4 * DD);
    convT(mlp_w2,  m2T,   LL, 4 * DD, DD);
    convT(cond_w1, cw1T,  1,  DD,     DD);
    convT(cond_w2, cw2T,  1,  DD,     DD);
    {
        int total = VPAD * DD;
        k_conv_pad<<<(total + 255) / 256, 256, 0, stream>>>(emb, lgT, DD, VV, total);
    }
    k_pad_cond<<<(256 * DD + 255) / 256, 256, 0, stream>>>(cond, condp);

    auto gemm = [&](const float* A, const unsigned short* WT, const float* bias,
                    const float* resid, float* C, int M, int N, int K,
                    int act, int logits) {
        dim3 grid((N + 63) / 64, (M + 255) / 256);
        k_gemm_bf16<<<grid, 256, 0, stream>>>(A, WT, bias, resid, C, M, N, K, act, logits);
    };

    // cond MLP: c = silu(cond @ w1 + b1) @ w2 + b2   (padded 256-row M, defined values)
    gemm(condp, cw1T, cond_b1, nullptr, chid, 256, DD, DD, ACT_SILU, 0);
    gemm(chid,  cw2T, cond_b2, nullptr, cout, 256, DD, DD, ACT_NONE, 0);

    k_embed<<<(MROWS * DD + 255) / 256, 256, 0, stream>>>(tokens, emb, pos_emb, cout, h);

    for (int i = 0; i < LL; ++i) {
        k_layernorm<<<MROWS, 256, 0, stream>>>(h, ln1_g + i * DD, ln1_b + i * DD, x);
        gemm(x, qkvT + (size_t)i * 3 * DD * DD, qkv_b + i * 3 * DD, nullptr,
             big, MROWS, 3 * DD, DD, ACT_NONE, 0);
        k_attention<<<BB * HH, 256, 0, stream>>>(big, attn);
        gemm(attn, projT + (size_t)i * DD * DD, proj_b + i * DD, h,
             h, MROWS, DD, DD, ACT_NONE, 0);
        k_layernorm<<<MROWS, 256, 0, stream>>>(h, ln2_g + i * DD, ln2_b + i * DD, x);
        gemm(x, m1T + (size_t)i * 4 * DD * DD, mlp_b1 + i * 4 * DD, nullptr,
             big, MROWS, 4 * DD, DD, ACT_GELU, 0);
        gemm(big, m2T + (size_t)i * DD * 4 * DD, mlp_b2 + i * DD, h,
             h, MROWS, DD, 4 * DD, ACT_NONE, 0);
    }

    k_layernorm<<<MROWS, 256, 0, stream>>>(h, lnf_g, lnf_b, x);
    gemm(x, lgT, nullptr, nullptr, outp, MROWS, VV, DD, ACT_NONE, 1);
}
